// Model_55774445306312
// MI455X (gfx1250) — compile-verified, matrix operand order checked by reading, not run on hardware
//
#include <hip/hip_runtime.h>
#include <math.h>

// ---------------- problem constants ----------------
#define cB 1024
#define cK 16
#define cL 32
#define cS 8
#define cD 256
#define cH 512
#define LOG2PI_F 1.8378770664093453f

typedef _Float16 v16h __attribute__((ext_vector_type(16)));
typedef _Float16 v8h  __attribute__((ext_vector_type(8)));
typedef float    v8f  __attribute__((ext_vector_type(8)));

// ---------------- small device helpers ----------------
__device__ __forceinline__ float softplusf(float x) {
  return (x > 20.f) ? x : log1pf(expf(x));
}
__device__ __forceinline__ unsigned pcg(unsigned x) {
  x = x * 747796405u + 2891336453u;
  unsigned w = ((x >> ((x >> 28) + 4u)) ^ x) * 277803737u;
  return (w >> 22) ^ w;
}
__device__ __forceinline__ float u01(unsigned h) {
  return ((float)(h >> 8) + 0.5f) * (1.0f / 16777216.0f);
}
__device__ __forceinline__ float rnormal(unsigned idx) {
  float u1 = u01(pcg(idx ^ 0xA511E9B3u));
  float u2 = u01(pcg(idx ^ 0x63D83595u));
  return sqrtf(-2.f * logf(u1)) * cosf(6.28318530718f * u2);
}

// CDNA5 async global->LDS 128-bit load (ASYNCcnt-tracked, no VGPR round trip).
// LDS destination address = low 32 bits of the generic pointer.
__device__ __forceinline__ void async_load_b128(const _Float16* gptr, _Float16* lptr) {
  unsigned lds = (unsigned)(unsigned long long)lptr;
  asm volatile("global_load_async_to_lds_b128 %0, %1, off"
               :: "v"(lds), "v"(gptr)
               : "memory");
}
__device__ __forceinline__ void wait_async0() {
  asm volatile("s_wait_asynccnt 0x0" ::: "memory");
}

// ---------------- f32 -> f16 converts ----------------
__global__ void f32_to_f16_kernel(const float* __restrict__ src,
                                  _Float16* __restrict__ dst, int n) {
  int i = blockIdx.x * blockDim.x + threadIdx.x;
  if (i < n) dst[i] = (_Float16)src[i];
}
// zero-pads columns [N, Npad) so GEMM tile loads need no bounds checks
__global__ void f32_to_f16_pad_kernel(const float* __restrict__ src,
                                      _Float16* __restrict__ dst,
                                      int K, int N, int Npad) {
  int i = blockIdx.x * blockDim.x + threadIdx.x;
  if (i < K * Npad) {
    int r = i / Npad, c = i % Npad;
    dst[i] = (c < N) ? (_Float16)src[(size_t)r * N + c] : (_Float16)0.f;
  }
}

// ---------------- WMMA GEMM: C = act(A(MxKd) @ B(KdxNld) + bias) ----------------
// Requirements (guaranteed by caller): M % 64 == 0, Kd % 32 == 0, Nld % 128 == 0.
// Stores guarded by n < N (true output width; row stride of outF/outH is N).
// block = 128 threads (4 waves); block tile 64x128; K-step 32; double-buffered LDS;
// A tile streamed with global_load_async_to_lds_b128, B tile pipelined via regs.
// act: 0=none 1=tanh 2=-0.5*softplus 3=softplus+1e-6
#define TM 64
#define TN 128
#define TKK 32
__global__ __launch_bounds__(128)
void gemm_wmma_kernel(const _Float16* __restrict__ A, const _Float16* __restrict__ Bm,
                      const float* __restrict__ bias,
                      float* __restrict__ outF, _Float16* __restrict__ outH,
                      int M, int N, int Nld, int Kd, int act) {
  __shared__ __align__(16) _Float16 As[2][TM][TKK + 8];    // row-major A tiles
  __shared__ __align__(16) _Float16 Bst[2][TN][TKK + 8];   // TRANSPOSED B tiles: [n][k]
  const int tid  = threadIdx.x;
  const int wave = tid >> 5;
  const int lane = tid & 31;
  const int half = lane >> 4;
  const int lr   = lane & 15;
  const int row0 = blockIdx.y * TM;
  const int col0 = blockIdx.x * TN;

  auto stageA = [&](int buf, int k0) {  // 2 async b128 per thread
#pragma unroll
    for (int it = 0; it < 2; it++) {
      int q = tid + it * 128;
      int r = q >> 2, c = (q & 3) * 8;
      async_load_b128(&A[(size_t)(row0 + r) * Kd + (k0 + c)], &As[buf][r][c]);
    }
  };
  auto loadBregs = [&](int k0, v8h* regs) {  // 4 vector global loads per thread
#pragma unroll
    for (int it = 0; it < 4; it++) {
      int q = tid + it * 128;
      int r = q >> 4, c = (q & 15) * 8;
      regs[it] = *(const v8h*)&Bm[(size_t)(k0 + r) * Nld + (col0 + c)];
    }
  };
  auto storeB = [&](int buf, const v8h* regs) {  // transposed scatter into LDS
#pragma unroll
    for (int it = 0; it < 4; it++) {
      int q = tid + it * 128;
      int r = q >> 4, c = (q & 15) * 8;
#pragma unroll
      for (int j = 0; j < 8; j++) Bst[buf][c + j][r] = regs[it][j];
    }
  };

  v8f acc[8];
#pragma unroll
  for (int i = 0; i < 8; i++) acc[i] = (v8f){0.f,0.f,0.f,0.f,0.f,0.f,0.f,0.f};

  // prologue: stage tile 0
  stageA(0, 0);
  {
    v8h regs[4];
    loadBregs(0, regs);
    storeB(0, regs);
  }
  wait_async0();
  __syncthreads();

  int cur = 0;
  for (int k0 = 0; k0 < Kd; k0 += TKK) {
    const int nb = cur ^ 1;
    const bool hasnext = (k0 + TKK) < Kd;
    v8h regs[4];
    if (hasnext) {
      stageA(nb, k0 + TKK);        // async A -> other buffer
      loadBregs(k0 + TKK, regs);   // B global loads issued before the math
    }

    // A fragment (ISA 7.12.2 16-bit A 16x32): row = wave*16 + lane%16,
    // kk = [half*8 .. +7] ++ [16+half*8 .. +7] -> two b128 LDS loads
    v16h afrag;
    {
      const v8h* pa = (const v8h*)&As[cur][wave * 16 + lr][0];
      v8h a0 = pa[half], a1 = pa[2 + half];
#pragma unroll
      for (int e = 0; e < 8; e++) { afrag[e] = a0[e]; afrag[8 + e] = a1[e]; }
    }
#pragma unroll
    for (int nt = 0; nt < 8; nt++) {
      const v8h* pb = (const v8h*)&Bst[cur][nt * 16 + lr][0];
      v8h b0 = pb[half], b1 = pb[2 + half];
      v16h bfrag;
#pragma unroll
      for (int e = 0; e < 8; e++) { bfrag[e] = b0[e]; bfrag[8 + e] = b1[e]; }
      acc[nt] = __builtin_amdgcn_wmma_f32_16x16x32_f16(
          false, afrag, false, bfrag, (short)0, acc[nt], false, false);
    }

    if (hasnext) storeB(nb, regs);  // loadcnt wait lands after the WMMA block
    wait_async0();
    __syncthreads();
    cur = nb;
  }

  // store: C/D layout: n = lane%16, m = r + 8*(lane/16)
#pragma unroll
  for (int nt = 0; nt < 8; nt++) {
#pragma unroll
    for (int r = 0; r < 8; r++) {
      int m = row0 + wave * 16 + half * 8 + r;
      int n = col0 + nt * 16 + lr;
      if (n < N) {
        float v = acc[nt][r] + (bias ? bias[n] : 0.f);
        if (act == 1)      v = tanhf(v);
        else if (act == 2) v = -0.5f * softplusf(v);
        else if (act == 3) v = softplusf(v) + 1e-6f;
        if (outF) outF[(size_t)m * N + n] = v;
        if (outH) outH[(size_t)m * N + n] = (_Float16)v;
      }
    }
  }
}

// ---------------- GMM natural params: eta2_g = -0.5 * Lk Lk^T ----------------
__global__ void gmm_nat_kernel(const float* __restrict__ phi_cov,
                               float* __restrict__ e2g) {
  int k = blockIdx.x, t = threadIdx.x;  // 256 threads
  __shared__ float Lk[cL][cL];
  for (int i = t; i < cL * cL; i += 256) {
    int r = i / cL, c = i % cL;
    float v = phi_cov[(size_t)k * cL * cL + i];
    if (c > r) v = 0.f;
    else if (c == r) v = softplusf(v);
    Lk[r][c] = v;
  }
  __syncthreads();
  for (int i = t; i < cL * cL; i += 256) {
    int r = i / cL, c = i % cL;
    float s = 0.f;
    for (int p = 0; p <= min(r, c); p++) s += Lk[r][p] * Lk[c][p];
    e2g[(size_t)k * cL * cL + i] = -0.5f * s;
  }
}

__global__ void logpi_kernel(const float* __restrict__ pi, float* __restrict__ lp) {
  if (threadIdx.x == 0) {
    float m = pi[0];
    for (int i = 1; i < cK; i++) m = fmaxf(m, pi[i]);
    float s = 0.f;
    for (int i = 0; i < cK; i++) s += expf(pi[i] - m);
    float lse = m + logf(s);
    for (int i = 0; i < cK; i++) lp[i] = pi[i] - lse;
  }
}

// ---------------- per-(b,k) dense 32x32 linear algebra + sampling ----------------
__global__ __launch_bounds__(32)
void bk_kernel(const float* __restrict__ e2g, const float* __restrict__ e2d,
               const float* __restrict__ eta1e, const float* __restrict__ phimu,
               const float* __restrict__ logpi,
               float* __restrict__ out_e1t, float* __restrict__ out_e2t,
               float* __restrict__ xk, _Float16* __restrict__ xk16,
               float* __restrict__ lp_out) {
  const int bk = blockIdx.x;
  const int b = bk / cK, k = bk % cK;
  const int t = threadIdx.x;
  __shared__ float A[cL * cL];   // -2*eta2_tilde -> Cholesky (lower)
  __shared__ float G[cL * cL];   // eta2_tilde
  __shared__ float X[cL * cL];   // A^{-1} * eta2_g
  __shared__ float M2[cL * cL];  // -2*w_eta2 -> Cholesky
  __shared__ float dv[cL], e1t[cL], mu[cL], w1[cL], meanv[cL], mt[cL], ts[cL];
  __shared__ float red[cL], scal[2];
  __shared__ float NZ[cL * cS];

  float dt = e2d[(size_t)b * cL + t];
  dv[t] = dt;
  float e1 = eta1e[(size_t)b * cL + t] + phimu[(size_t)k * cL + t];
  e1t[t] = e1;
  out_e1t[(size_t)bk * cL + t] = e1;
  mu[t] = eta1e[(size_t)b * cL + t] / (-2.f * dt);
  for (int i = 0; i < cL; i++) {
    float g = e2g[((size_t)k * cL + i) * cL + t];
    float et = g + ((i == t) ? dt : 0.f);
    G[i * cL + t] = et;
    out_e2t[((size_t)bk * cL + i) * cL + t] = et;
    A[i * cL + t] = -2.f * et;
  }
  __syncthreads();

  // Cholesky of A
  for (int j = 0; j < cL; j++) {
    if (t == 0) {
      float s = A[j * cL + j];
      for (int p = 0; p < j; p++) { float v = A[j * cL + p]; s -= v * v; }
      A[j * cL + j] = sqrtf(fmaxf(s, 1e-20f));
    }
    __syncthreads();
    float dj = A[j * cL + j];
    if (t > j) {
      float s = A[t * cL + j];
      for (int p = 0; p < j; p++) s -= A[t * cL + p] * A[j * cL + p];
      A[t * cL + j] = s / dj;
    }
    __syncthreads();
  }

  // X column t: A^{-1} * eta2_g[:,t]
  for (int i = 0; i < cL; i++) {
    float s = G[i * cL + t] - ((i == t) ? dv[t] : 0.f);
    for (int p = 0; p < i; p++) s -= A[i * cL + p] * X[p * cL + t];
    X[i * cL + t] = s / A[i * cL + i];
  }
  for (int i = cL - 1; i >= 0; i--) {
    float s = X[i * cL + t];
    for (int p = i + 1; p < cL; p++) s -= A[p * cL + i] * X[p * cL + t];
    X[i * cL + t] = s / A[i * cL + i];
  }
  __syncthreads();

  // parallel: t0 -> mean solve; t1 -> mu_t solve; t2..9 -> noise back-subs
  if (t == 0) {
    for (int i = 0; i < cL; i++) {
      float s = e1t[i];
      for (int p = 0; p < i; p++) s -= A[i * cL + p] * meanv[p];
      meanv[i] = s / A[i * cL + i];
    }
    for (int i = cL - 1; i >= 0; i--) {
      float s = meanv[i];
      for (int p = i + 1; p < cL; p++) s -= A[p * cL + i] * meanv[p];
      meanv[i] = s / A[i * cL + i];
    }
  } else if (t == 1) {
    for (int i = 0; i < cL; i++) {
      float s = phimu[(size_t)k * cL + i];
      for (int p = 0; p < i; p++) s -= A[i * cL + p] * mt[p];
      mt[i] = s / A[i * cL + i];
    }
    for (int i = cL - 1; i >= 0; i--) {
      float s = mt[i];
      for (int p = i + 1; p < cL; p++) s -= A[p * cL + i] * mt[p];
      mt[i] = s / A[i * cL + i];
    }
  } else if (t >= 2 && t < 2 + cS) {
    int s = t - 2;
    for (int i = cL - 1; i >= 0; i--) {
      float e = rnormal(((unsigned)bk * cL + (unsigned)i) * cS + (unsigned)s);
      for (int p = i + 1; p < cL; p++) e -= A[p * cL + i] * NZ[p * cS + s];
      NZ[i * cS + s] = e / A[i * cL + i];
    }
  }
  __syncthreads();

  w1[t] = -2.f * dv[t] * mt[t];
  for (int i = 0; i < cL; i++)
    M2[i * cL + t] = 2.f * (dv[t] * X[t * cL + i] + dv[i] * X[i * cL + t]);
  __syncthreads();

  {
    float s = 0.f;
    for (int i = 0; i < cL; i++) s += mu[i] * M2[i * cL + t];
    red[t] = mu[t] * w1[t] - 0.5f * mu[t] * s;
  }
  __syncthreads();
  if (t == 0) { float s = 0.f; for (int i = 0; i < cL; i++) s += red[i]; scal[0] = s; }
  __syncthreads();

  // Cholesky of M2
  for (int j = 0; j < cL; j++) {
    if (t == 0) {
      float s = M2[j * cL + j];
      for (int p = 0; p < j; p++) { float v = M2[j * cL + p]; s -= v * v; }
      M2[j * cL + j] = sqrtf(fmaxf(s, 1e-20f));
    }
    __syncthreads();
    float dj = M2[j * cL + j];
    if (t > j) {
      float s = M2[t * cL + j];
      for (int p = 0; p < j; p++) s -= M2[t * cL + p] * M2[j * cL + p];
      M2[t * cL + j] = s / dj;
    }
    __syncthreads();
  }
  red[t] = logf(fmaxf(M2[t * cL + t], 1e-30f));
  __syncthreads();
  if (t == 0) {
    float ld = 0.f;
    for (int i = 0; i < cL; i++) ld += red[i];
    for (int i = 0; i < cL; i++) {
      float s = w1[i];
      for (int p = 0; p < i; p++) s -= M2[i * cL + p] * ts[p];
      ts[i] = s / M2[i * cL + i];
    }
    for (int i = cL - 1; i >= 0; i--) {
      float s = ts[i];
      for (int p = i + 1; p < cL; p++) s -= M2[p * cL + i] * ts[p];
      ts[i] = s / M2[i * cL + i];
    }
    float st = 0.f;
    for (int i = 0; i < cL; i++) st += w1[i] * ts[i];
    lp_out[bk] = scal[0] + ld - 0.5f * st - 0.5f * cL * LOG2PI_F + logpi[k];
  }
  __syncthreads();

  for (int s = 0; s < cS; s++) {
    float v = meanv[t] + NZ[t * cS + s];
    size_t o = ((size_t)bk * cS + s) * cL + t;
    xk[o] = v;
    xk16[o] = (_Float16)v;
  }
}

// ---------------- log_softmax over K, categorical z, gather x_samples ----------------
__global__ __launch_bounds__(32)
void finalize_kernel(const float* __restrict__ lpbuf, const float* __restrict__ xk,
                     float* __restrict__ log_z, float* __restrict__ x_samples) {
  int b = blockIdx.x, t = threadIdx.x;
  __shared__ float sh[cK];
  __shared__ int zsel;
  if (t < cK) sh[t] = lpbuf[(size_t)b * cK + t];
  __syncthreads();
  if (t == 0) {
    float m = sh[0];
    for (int i = 1; i < cK; i++) m = fmaxf(m, sh[i]);
    float s = 0.f;
    for (int i = 0; i < cK; i++) s += expf(sh[i] - m);
    float lse = m + logf(s);
    for (int i = 0; i < cK; i++) sh[i] -= lse;
    float u = u01(pcg(0x9E3779B9u + (unsigned)b));
    float c = 0.f; int z = cK - 1;
    for (int i = 0; i < cK; i++) { c += expf(sh[i]); if (u < c) { z = i; break; } }
    zsel = z;
  }
  __syncthreads();
  if (t < cK) log_z[(size_t)b * cK + t] = sh[t];
  x_samples[(size_t)b * cL + t] = xk[(((size_t)b * cK + zsel) * cS + 0) * cL + t];
}

// ---------------- host-side orchestration ----------------
extern "C" void kernel_launch(void* const* d_in, const int* in_sizes, int n_in,
                              void* d_out, int out_size, void* d_ws, size_t ws_size,
                              hipStream_t stream) {
  const float* y        = (const float*)d_in[0];
  const float* enc_w1   = (const float*)d_in[1];
  const float* enc_b1   = (const float*)d_in[2];
  const float* enc_w2   = (const float*)d_in[3];
  const float* enc_b2   = (const float*)d_in[4];
  const float* enc_we1  = (const float*)d_in[5];
  const float* enc_be1  = (const float*)d_in[6];
  const float* enc_we2  = (const float*)d_in[7];
  const float* enc_be2  = (const float*)d_in[8];
  const float* phi_mu   = (const float*)d_in[9];
  const float* phi_cov  = (const float*)d_in[10];
  const float* train_pi = (const float*)d_in[11];
  const float* dec_w1   = (const float*)d_in[12];
  const float* dec_b1   = (const float*)d_in[13];
  const float* dec_w2   = (const float*)d_in[14];
  const float* dec_b2   = (const float*)d_in[15];
  const float* dec_wm   = (const float*)d_in[16];
  const float* dec_bm   = (const float*)d_in[17];
  const float* dec_wv   = (const float*)d_in[18];
  const float* dec_bv   = (const float*)d_in[19];

  float* out = (float*)d_out;
  const size_t o_ymean = 0;
  const size_t o_yvar  = o_ymean + (size_t)cB * cK * cS * cD;
  const size_t o_xk    = o_yvar + (size_t)cB * cK * cS * cD;
  const size_t o_xs    = o_xk + (size_t)cB * cK * cS * cL;
  const size_t o_logz  = o_xs + (size_t)cB * cL;
  const size_t o_e1t   = o_logz + (size_t)cB * cK;
  const size_t o_e2t   = o_e1t + (size_t)cB * cK * cL;

  char* wp = (char*)d_ws;
  auto alloc = [&](size_t bytes) -> void* {
    void* p = (void*)wp;
    wp += (bytes + 255) & ~(size_t)255;
    return p;
  };
  const int NpadL = 128;  // pad 32-col weight matrices to one full N-tile
  _Float16* w1_16  = (_Float16*)alloc((size_t)cD * cH * 2);
  _Float16* w2_16  = (_Float16*)alloc((size_t)cH * cH * 2);
  _Float16* we1_16 = (_Float16*)alloc((size_t)cH * NpadL * 2);
  _Float16* we2_16 = (_Float16*)alloc((size_t)cH * NpadL * 2);
  _Float16* dw1_16 = (_Float16*)alloc((size_t)cL * cH * 2);
  _Float16* dw2_16 = (_Float16*)alloc((size_t)cH * cH * 2);
  _Float16* dwm_16 = (_Float16*)alloc((size_t)cH * cD * 2);
  _Float16* dwv_16 = (_Float16*)alloc((size_t)cH * cD * 2);
  _Float16* y16    = (_Float16*)alloc((size_t)cB * cD * 2);
  _Float16* h1e    = (_Float16*)alloc((size_t)cB * cH * 2);
  _Float16* h2e    = (_Float16*)alloc((size_t)cB * cH * 2);
  float* eta1e  = (float*)alloc((size_t)cB * cL * 4);
  float* e2d    = (float*)alloc((size_t)cB * cL * 4);
  float* e2g    = (float*)alloc((size_t)cK * cL * cL * 4);
  float* lpi    = (float*)alloc(cK * 4);
  float* lp     = (float*)alloc((size_t)cB * cK * 4);
  _Float16* xk16 = (_Float16*)alloc((size_t)cB * cK * cS * cL * 2);
  const int MC = 16384;  // decoder M-chunk
  _Float16* h1d = (_Float16*)alloc((size_t)MC * cH * 2);
  _Float16* h2d = (_Float16*)alloc((size_t)MC * cH * 2);

  auto conv = [&](const float* s, _Float16* dd, int n) {
    f32_to_f16_kernel<<<(n + 255) / 256, 256, 0, stream>>>(s, dd, n);
  };
  auto convpad = [&](const float* s, _Float16* dd, int Kr, int N, int Npad) {
    int n = Kr * Npad;
    f32_to_f16_pad_kernel<<<(n + 255) / 256, 256, 0, stream>>>(s, dd, Kr, N, Npad);
  };
  auto gemm = [&](const _Float16* A, const _Float16* Bm, const float* bias,
                  float* oF, _Float16* oH, int M, int N, int Nld, int Kd, int act) {
    dim3 g(Nld / TN, M / TM), blk(128);
    gemm_wmma_kernel<<<g, blk, 0, stream>>>(A, Bm, bias, oF, oH, M, N, Nld, Kd, act);
  };

  // weight + input conversion (f16, padded where N==32)
  conv(enc_w1, w1_16, cD * cH);
  conv(enc_w2, w2_16, cH * cH);
  convpad(enc_we1, we1_16, cH, cL, NpadL);
  convpad(enc_we2, we2_16, cH, cL, NpadL);
  conv(dec_w1, dw1_16, cL * cH);
  conv(dec_w2, dw2_16, cH * cH);
  conv(dec_wm, dwm_16, cH * cD);
  conv(dec_wv, dwv_16, cH * cD);
  conv(y, y16, cB * cD);

  // GMM natural params
  logpi_kernel<<<1, 32, 0, stream>>>(train_pi, lpi);
  gmm_nat_kernel<<<cK, 256, 0, stream>>>(phi_cov, e2g);

  // encoder (WMMA GEMMs)
  gemm(y16, w1_16, enc_b1, nullptr, h1e, cB, cH, cH, cD, 1);
  gemm(h1e, w2_16, enc_b2, nullptr, h2e, cB, cH, cH, cH, 1);
  gemm(h2e, we1_16, enc_be1, eta1e, nullptr, cB, cL, NpadL, cH, 0);
  gemm(h2e, we2_16, enc_be2, e2d, nullptr, cB, cL, NpadL, cH, 2);

  // per-(b,k) dense algebra: eta tilde, log_z logits, latent samples
  bk_kernel<<<cB * cK, 32, 0, stream>>>(e2g, e2d, eta1e, phi_mu, lpi,
                                        out + o_e1t, out + o_e2t,
                                        out + o_xk, xk16, lp);

  // log_z + categorical + gather
  finalize_kernel<<<cB, 32, 0, stream>>>(lp, out + o_xk, out + o_logz, out + o_xs);

  // decoder (WMMA GEMMs, chunked over M = B*K*S = 131072)
  const int Mtot = cB * cK * cS;
  for (int c = 0; c < Mtot / MC; c++) {
    const _Float16* xa = xk16 + (size_t)c * MC * cL;
    gemm(xa, dw1_16, dec_b1, nullptr, h1d, MC, cH, cH, cL, 1);
    gemm(h1d, dw2_16, dec_b2, nullptr, h2d, MC, cH, cH, cH, 1);
    gemm(h2d, dwm_16, dec_bm, out + o_ymean + (size_t)c * MC * cD, nullptr, MC, cD, cD, cH, 0);
    gemm(h2d, dwv_16, dec_bv, out + o_yvar + (size_t)c * MC * cD, nullptr, MC, cD, cD, cH, 3);
  }
  (void)in_sizes; (void)n_in; (void)out_size; (void)ws_size;
}
